// GraphSetAE_41334765257102
// MI455X (gfx1250) — matrix operand order repaired and checked by reading, not compiled, not run on hardware
//
#include <hip/hip_runtime.h>
#include <math.h>

// ---------------------------------------------------------------------------
// Types for CDNA5 WMMA
// ---------------------------------------------------------------------------
typedef __attribute__((ext_vector_type(16))) __bf16 v16bf;
typedef __attribute__((ext_vector_type(8)))  float  v8f;
typedef __attribute__((ext_vector_type(2)))  float  v2f;
typedef __attribute__((ext_vector_type(4)))  unsigned v4u;

union FragAB { v16bf bf; v4u u[2]; };

__device__ __forceinline__ unsigned short f2bf(float f) {
    unsigned u = __float_as_uint(f);
    unsigned r = (u + 0x7FFFu + ((u >> 16) & 1u)) >> 16;   // RNE
    return (unsigned short)r;
}

__device__ __forceinline__ float gelu_f(float x) {
    float x3 = x * x * x;
    return 0.5f * x * (1.f + tanhf(0.7978845608028654f * (x + 0.044715f * x3)));
}

// ---------------------------------------------------------------------------
// Generic WMMA bf16 GEMM:  C(MxN) = A(MxK) * B(KxN), row-major fp32 in/out.
// 256 threads = 8 wave32s. Block tile BM x BN, K-chunks of 32.
// LDS tiles are stored PRE-SWIZZLED into the ISA fragment layout so each lane
// fetches its fragment as two contiguous b128 LDS reads.
//   A 16x32 bf16: lane L (m=L%15? m=L&15): L<16 -> K {0..7,16..23}, L>=16 -> K {8..15,24..31}
//   B 32x16 bf16: lane L: n=L&15, L<16 -> K {0..7,16..23}, L>=16 -> K {8..15,24..31}
// ---------------------------------------------------------------------------
template <int BM, int BN>
__global__ __launch_bounds__(256)
void wmma_gemm_kernel(const float* __restrict__ A, const float* __restrict__ B,
                      float* __restrict__ C, int M, int N, int K)
{
    constexpr int TM = BM / 16, TN = BN / 16;
    constexpr int TILES = TM * TN;
    constexpr int TPW = TILES / 8;        // tiles per wave
    constexpr int KC = 32;

    __shared__ __align__(16) unsigned short sA[TM * 512];
    __shared__ __align__(16) unsigned short sB[TN * 512];

    const int tid  = threadIdx.x;
    const int wave = tid >> 5;
    const int lane = tid & 31;
    const int m0 = blockIdx.x * BM;
    const int n0 = blockIdx.y * BN;

    v8f zero = {};
    v8f acc[TPW];
#pragma unroll
    for (int t = 0; t < TPW; ++t) acc[t] = zero;

    for (int k0 = 0; k0 < K; k0 += KC) {
        // ---- stage A (BM x 32), swizzled into fragment order ----
#pragma unroll
        for (int e = tid; e < BM * KC; e += 256) {
            int r = e >> 5;                 // row within block tile
            int k = e & 31;
            int gm = m0 + r, gk = k0 + k;
            float f = (gm < M && gk < K) ? A[(size_t)gm * K + gk] : 0.f;
            if (gk + KC < K && gm < M)      // hint next chunk of the A stream
                __builtin_prefetch(&A[(size_t)gm * K + gk + KC], 0, 1);
            int tm = r >> 4, lrow = r & 15;
            int lane_s = lrow + 16 * ((k >> 3) & 1);
            int j = (k & 7) + 8 * (k >> 4);
            sA[tm * 512 + lane_s * 16 + j] = f2bf(f);
        }
        // ---- stage B (32 x BN), swizzled into fragment order ----
#pragma unroll
        for (int e = tid; e < KC * BN; e += 256) {
            int k = e / BN;
            int n = e % BN;
            int gk = k0 + k, gn = n0 + n;
            float f = (gk < K && gn < N) ? B[(size_t)gk * N + gn] : 0.f;
            int tn = n >> 4, lcol = n & 15;
            int lane_s = lcol + 16 * ((k >> 3) & 1);
            int j = (k & 7) + 8 * (k >> 4);
            sB[tn * 512 + lane_s * 16 + j] = f2bf(f);
        }
        __syncthreads();

#pragma unroll
        for (int t = 0; t < TPW; ++t) {
            int tile = wave * TPW + t;
            int tm = tile / TN, tn = tile % TN;
            FragAB fa, fb;
            const v4u* pa = reinterpret_cast<const v4u*>(&sA[tm * 512 + lane * 16]);
            fa.u[0] = pa[0]; fa.u[1] = pa[1];
            const v4u* pb = reinterpret_cast<const v4u*>(&sB[tn * 512 + lane * 16]);
            fb.u[0] = pb[0]; fb.u[1] = pb[1];
            acc[t] = __builtin_amdgcn_wmma_f32_16x16x32_bf16(
                false, fa.bf, false, fb.bf, (short)0, acc[t], false, false);
        }
        __syncthreads();
    }

    // ---- epilogue: C layout lanes0-15 M=r, lanes16-31 M=r+8, N=lane&15 ----
#pragma unroll
    for (int t = 0; t < TPW; ++t) {
        int tile = wave * TPW + t;
        int tm = tile / TN, tn = tile % TN;
        int mbase = m0 + tm * 16 + 8 * (lane >> 4);
        int ncol  = n0 + tn * 16 + (lane & 15);
        if (ncol < N) {
#pragma unroll
            for (int r = 0; r < 8; ++r) {
                int gm = mbase + r;
                if (gm < M) C[(size_t)gm * N + ncol] = acc[t][r];
            }
        }
    }
}

// ---------------------------------------------------------------------------
// Per-row (H=256) optional-bias + LayerNorm + GELU. One wave32 per row.
// ---------------------------------------------------------------------------
__global__ __launch_bounds__(256)
void ln_gelu_kernel(const float* __restrict__ x, const float* __restrict__ bias,
                    const float* __restrict__ g, const float* __restrict__ b,
                    float* __restrict__ y, int rows)
{
    const int H = 256;
    int wave = threadIdx.x >> 5, lane = threadIdx.x & 31;
    int row = blockIdx.x * 8 + wave;
    if (row >= rows) return;
    float v[8], s = 0.f, sq = 0.f;
#pragma unroll
    for (int j = 0; j < 8; ++j) {
        int col = lane + 32 * j;
        float t = x[row * H + col];
        if (bias) t += bias[col];
        v[j] = t; s += t; sq += t * t;
    }
#pragma unroll
    for (int off = 16; off >= 1; off >>= 1) {
        s  += __shfl_xor(s,  off, 32);
        sq += __shfl_xor(sq, off, 32);
    }
    float mean = s * (1.f / 256.f);
    float var  = sq * (1.f / 256.f) - mean * mean;
    float rstd = rsqrtf(var + 1e-5f);
#pragma unroll
    for (int j = 0; j < 8; ++j) {
        int col = lane + 32 * j;
        float t = (v[j] - mean) * rstd * g[col] + b[col];
        y[row * H + col] = gelu_f(t);
    }
}

// ---------------------------------------------------------------------------
// GAT attention per graph b: hh (16x256, = 16 x [4 heads x 64]) -> out (16x256)
// ---------------------------------------------------------------------------
__global__ __launch_bounds__(128)
void gat_attn_kernel(const float* __restrict__ hh, const float* __restrict__ a_src_w,
                     const float* __restrict__ a_dst_w, const float* __restrict__ adj,
                     float* __restrict__ out)
{
    __shared__ float sh[16 * 256];
    __shared__ float s_as[64], s_ad[64];
    int b = blockIdx.x, tid = threadIdx.x;
    const float* hb = hh + (size_t)b * 16 * 256;
    for (int e = tid; e < 16 * 256; e += 128) sh[e] = hb[e];
    __syncthreads();
    if (tid < 64) {
        int s = tid >> 2, hd = tid & 3;
        float as = 0.f, ad = 0.f;
        for (int d = 0; d < 64; ++d) {
            float v = sh[s * 256 + hd * 64 + d];
            as += v * a_src_w[hd * 64 + d];
            ad += v * a_dst_w[hd * 64 + d];
        }
        s_as[tid] = as; s_ad[tid] = ad;   // index = s*4 + head
    }
    __syncthreads();
    if (tid < 64) {
        int i = tid >> 2, hd = tid & 3;
        float e[16], mx = -3.4e38f;
        float ai = s_as[i * 4 + hd];
        for (int j = 0; j < 16; ++j) {
            float t = ai + s_ad[j * 4 + hd];
            t = t > 0.f ? t : 0.2f * t;                    // leaky_relu 0.2
            if (adj[b * 256 + i * 16 + j] == 0.f) t = -3.4e38f;
            e[j] = t; mx = fmaxf(mx, t);
        }
        float sum = 0.f;
        for (int j = 0; j < 16; ++j) { e[j] = __expf(e[j] - mx); sum += e[j]; }
        float inv = 1.f / sum;
        for (int d = 0; d < 64; ++d) {
            float acc = 0.f;
            for (int j = 0; j < 16; ++j) acc += e[j] * sh[j * 256 + hd * 64 + d];
            out[((size_t)b * 16 + i) * 256 + hd * 64 + d] = acc * inv;
        }
    }
}

// ---------------------------------------------------------------------------
// Elementwise bias (+optional GELU)
// ---------------------------------------------------------------------------
__global__ void bias_act_kernel(const float* __restrict__ x, const float* __restrict__ bias,
                                float* __restrict__ y, int n, int ncols, int act)
{
    int i = blockIdx.x * blockDim.x + threadIdx.x;
    if (i >= n) return;
    float t = x[i] + bias[i % ncols];
    y[i] = act ? gelu_f(t) : t;
}

__global__ void gather_center_kernel(const float* __restrict__ z_all,
                                     const int* __restrict__ center,
                                     float* __restrict__ z_center)
{
    int i = blockIdx.x * blockDim.x + threadIdx.x;
    if (i >= 64 * 64) return;
    int b = i >> 6, d = i & 63;
    z_center[i] = z_all[((size_t)b * 16 + *center) * 64 + d];
}

// ---------------------------------------------------------------------------
// Row softmax over G -> mu = lib * rho ; theta = exp(log_theta) broadcast
// ---------------------------------------------------------------------------
__global__ __launch_bounds__(256)
void softmax_mu_theta_kernel(const float* __restrict__ logits, const float* __restrict__ bo,
                             const float* __restrict__ library, const int* __restrict__ center,
                             const float* __restrict__ log_theta,
                             float* __restrict__ mu, float* __restrict__ theta, int G, int S)
{
    __shared__ float red[256];
    int b = blockIdx.x, tid = threadIdx.x;
    const float* lr = logits + (size_t)b * G;
    float mx = -3.4e38f;
    for (int g = tid; g < G; g += 256) mx = fmaxf(mx, lr[g] + bo[g]);
    red[tid] = mx; __syncthreads();
    for (int o = 128; o >= 1; o >>= 1) { if (tid < o) red[tid] = fmaxf(red[tid], red[tid + o]); __syncthreads(); }
    mx = red[0]; __syncthreads();
    float s = 0.f;
    for (int g = tid; g < G; g += 256) s += __expf(lr[g] + bo[g] - mx);
    red[tid] = s; __syncthreads();
    for (int o = 128; o >= 1; o >>= 1) { if (tid < o) red[tid] += red[tid + o]; __syncthreads(); }
    float inv = 1.f / red[0];
    float lib = library[b * S + *center];
    for (int g = tid; g < G; g += 256) {
        float rho = __expf(lr[g] + bo[g] - mx) * inv;
        mu[(size_t)b * G + g]    = lib * rho;
        theta[(size_t)b * G + g] = __expf(log_theta[g]);
    }
}

// ---------------------------------------------------------------------------
// Per graph: normalize rows of z (16x64), emit z_all, and adj_pred = zn*zn^T
// via V_WMMA_F32_16X16X4_F32 (A and B fragments are identical for the
// symmetric product given the ISA 16x4 / 4x16 f32 layouts). One wave32/graph.
// ---------------------------------------------------------------------------
__global__ __launch_bounds__(32)
void znorm_adjpred_kernel(const float* __restrict__ z_all, float* __restrict__ z_out,
                          float* __restrict__ adj_pred)
{
    __shared__ float zn[16 * 64];
    int b = blockIdx.x, lane = threadIdx.x;
    const float* zb = z_all + (size_t)b * 16 * 64;
    for (int e = lane; e < 1024; e += 32) {
        float v = zb[e];
        zn[e] = v;
        z_out[(size_t)b * 1024 + e] = v;
    }
    __syncthreads();
    int r  = lane & 15;
    int c0 = (lane >> 4) * 32;
    float ss = 0.f;
    for (int c = 0; c < 32; ++c) { float v = zn[r * 64 + c0 + c]; ss += v * v; }
    ss += __shfl_xor(ss, 16, 32);
    float inv = 1.f / fmaxf(sqrtf(ss), 1e-12f);
    for (int c = 0; c < 32; ++c) zn[r * 64 + c0 + c] *= inv;
    __syncthreads();

    v8f acc = {};
    int mrow = lane & 15;
    int koff = (lane >> 4) * 2;   // lanes 0-15: K k0+0/1 ; lanes 16-31: K k0+2/3
#pragma unroll
    for (int k0 = 0; k0 < 64; k0 += 4) {
        v2f a;
        a.x = zn[mrow * 64 + k0 + koff];
        a.y = zn[mrow * 64 + k0 + koff + 1];
        acc = __builtin_amdgcn_wmma_f32_16x16x4_f32(
            false, a, false, a, (short)0, acc, false, false);
    }
    int n = lane & 15;
    int mbase = 8 * (lane >> 4);
#pragma unroll
    for (int rr = 0; rr < 8; ++rr)
        adj_pred[(size_t)b * 256 + (mbase + rr) * 16 + n] = acc[rr];
}

// ---------------------------------------------------------------------------
// Launch
// ---------------------------------------------------------------------------
extern "C" void kernel_launch(void* const* d_in, const int* in_sizes, int n_in,
                              void* d_out, int out_size, void* d_ws, size_t ws_size,
                              hipStream_t stream)
{
    const float* x_set     = (const float*)d_in[0];
    const float* library   = (const float*)d_in[1];
    const float* adj       = (const float*)d_in[2];
    const float* w_in      = (const float*)d_in[3];
    const float* b_in      = (const float*)d_in[4];
    const float* ln_in_g   = (const float*)d_in[5];
    const float* ln_in_b   = (const float*)d_in[6];
    const float* gat_W     = (const float*)d_in[7];
    const float* gat_a_src = (const float*)d_in[8];
    const float* gat_a_dst = (const float*)d_in[9];
    const float* gat_ln_g  = (const float*)d_in[10];
    const float* gat_ln_b  = (const float*)d_in[11];
    const float* zw        = (const float*)d_in[12];
    const float* zb        = (const float*)d_in[13];
    const float* dec_w1    = (const float*)d_in[14];
    const float* dec_b1    = (const float*)d_in[15];
    const float* dec_w2    = (const float*)d_in[16];
    const float* dec_b2    = (const float*)d_in[17];
    const float* dec_wo    = (const float*)d_in[18];
    const float* dec_bo    = (const float*)d_in[19];
    const float* log_theta = (const float*)d_in[20];
    const int*   center    = (const int*)d_in[21];

    const int B = 64, S = 16, G = 28231, H = 256, L = 64;
    const int BS = B * S;                       // 1024

    float* ws   = (float*)d_ws;
    float* bufA = ws;                           // 1024x256 raw GEMM out / GAT out
    float* bufB = bufA + 262144;                // 1024x256 current h
    float* bufC = bufB + 262144;                // 1024x256 hh
    float* bufD = bufC + 262144;                // 1024x64  z_all
    float* bufE = bufD + 65536;                 // 64x64    z_center
    float* bufF = bufE + 4096;                  // 64x256   d1
    float* bufG = bufF + 16384;                 // 64x256   d2
    float* bufH = bufG + 16384;                 // 64x28231 logits

    float* mu    = (float*)d_out;
    float* theta = mu + (size_t)B * G;
    float* zout  = theta + (size_t)B * G;
    float* adjp  = zout + (size_t)B * S * L;

    // 1) h = gelu(ln(x_set @ w_in + b_in))  -- HBM-bound: A read once, 64 CTAs
    wmma_gemm_kernel<16, 256><<<dim3(64, 1), 256, 0, stream>>>(x_set, w_in, bufA, BS, H, G);
    ln_gelu_kernel<<<128, 256, 0, stream>>>(bufA, b_in, ln_in_g, ln_in_b, bufB, BS);

    // 2) two GAT layers
    for (int i = 0; i < 2; ++i) {
        wmma_gemm_kernel<64, 64><<<dim3(16, 4), 256, 0, stream>>>(
            bufB, gat_W + (size_t)i * H * H, bufC, BS, H, H);
        gat_attn_kernel<<<64, 128, 0, stream>>>(
            bufC, gat_a_src + i * 256, gat_a_dst + i * 256, adj, bufA);
        ln_gelu_kernel<<<128, 256, 0, stream>>>(
            bufA, nullptr, gat_ln_g + i * H, gat_ln_b + i * H, bufB, BS);
    }

    // 3) z_all = h @ zw + zb
    wmma_gemm_kernel<64, 64><<<dim3(16, 1), 256, 0, stream>>>(bufB, zw, bufD, BS, L, H);
    bias_act_kernel<<<(BS * L + 255) / 256, 256, 0, stream>>>(bufD, zb, bufD, BS * L, L, 0);

    // 4) decoder on z_center
    gather_center_kernel<<<16, 256, 0, stream>>>(bufD, center, bufE);
    wmma_gemm_kernel<64, 64><<<dim3(1, 4), 256, 0, stream>>>(bufE, dec_w1, bufF, B, H, L);
    bias_act_kernel<<<64, 256, 0, stream>>>(bufF, dec_b1, bufF, B * H, H, 1);
    wmma_gemm_kernel<64, 64><<<dim3(1, 4), 256, 0, stream>>>(bufF, dec_w2, bufG, B, H, H);
    bias_act_kernel<<<64, 256, 0, stream>>>(bufG, dec_b2, bufG, B * H, H, 1);
    wmma_gemm_kernel<64, 64><<<dim3(1, 442), 256, 0, stream>>>(bufG, dec_wo, bufH, B, G, H);

    // 5) softmax -> mu, theta
    softmax_mu_theta_kernel<<<64, 256, 0, stream>>>(bufH, dec_bo, library, center,
                                                    log_theta, mu, theta, G, S);

    // 6) z_all out + adj_pred = zn @ zn^T (f32 WMMA)
    znorm_adjpred_kernel<<<64, 32, 0, stream>>>(bufD, zout, adjp);
}
